// Layer_67637144978060
// MI455X (gfx1250) — matrix operand order repaired and checked by reading, not compiled, not run on hardware
//
#include <hip/hip_runtime.h>

// ---------------------------------------------------------------------------
// MI455X (gfx1250, wave32). Compute-bound (~412 GFLOP vs ~300MB HBM)
// -> bf16 WMMA 16x16x32 with f32 accumulation.
// Generic LDS-staged GEMM: macro tile 128x128, 8 waves, wave tile 32x64
// (2x4 WMMA tiles). B tile stored TRANSPOSED in LDS so every fragment
// gather is a 16B-aligned ds_load_b128 (no scalar ds_load_u16 / v_mov_b16).
// ---------------------------------------------------------------------------

typedef __attribute__((ext_vector_type(16))) __bf16          v16bf;
typedef __attribute__((ext_vector_type(16))) unsigned short  v16u;
typedef __attribute__((ext_vector_type(8)))  unsigned short  v8u;
typedef __attribute__((ext_vector_type(8)))  float           v8f;

static constexpr int BM = 128;       // macro tile M
static constexpr int BN = 128;       // macro tile N
static constexpr int BK = 32;        // K step (= WMMA K)
static constexpr int LDP = BK + 8;   // row pitch in ushorts: 80B = 5*16B
static constexpr int NTHREADS = 256; // 8 waves (wave32)

#define EPS_F 1e-6f

__device__ __forceinline__ unsigned short f2bf(float f) {
  unsigned int u = __float_as_uint(f);
  unsigned int r = (u + 0x7FFFu + ((u >> 16) & 1u)) >> 16;  // RNE
  return (unsigned short)r;
}
__device__ __forceinline__ float bf2f(unsigned short h) {
  return __uint_as_float(((unsigned int)h) << 16);
}

// ---------------------------------------------------------------------------
// Elementwise / setup kernels
// ---------------------------------------------------------------------------

__global__ void compute_lg_kernel(const float* __restrict__ decay,
                                  float* __restrict__ lg) {
  if (threadIdx.x == 0 && blockIdx.x == 0) {
    lg[0] = -log1pf(__expf(-decay[0]));   // log(sigmoid(d))
  }
}

__global__ __launch_bounds__(256)
void f32_to_bf16_kernel(const float* __restrict__ src,
                        unsigned short* __restrict__ dst, long long n) {
  long long i = (long long)blockIdx.x * blockDim.x + threadIdx.x;
  if (i < n) dst[i] = f2bf(src[i]);
}

__global__ __launch_bounds__(256)
void rmsnorm_kernel(const float* __restrict__ x, const float* __restrict__ g,
                    float* __restrict__ y_partial,
                    unsigned short* __restrict__ out_bf,
                    unsigned short* __restrict__ wk_bf, int T, int D) {
  long long row = blockIdx.x;               // b*T + t
  int t = (int)(row % T);
  const float* xr = x + row * (long long)D;
  __shared__ float red[256];
  float ss = 0.f;
  for (int i = threadIdx.x; i < D; i += 256) { float v = xr[i]; ss += v * v; }
  red[threadIdx.x] = ss;
  __syncthreads();
  for (int off = 128; off > 0; off >>= 1) {
    if ((int)threadIdx.x < off) red[threadIdx.x] += red[threadIdx.x + off];
    __syncthreads();
  }
  float inv = rsqrtf(red[0] / (float)D + EPS_F);
  long long base = row * (long long)D;
  for (int i = threadIdx.x; i < D; i += 256) {
    float o = xr[i] * inv * g[i];
    y_partial[base + i] = xr[i] + o;        // y = x + out (+0.03*proj later)
    out_bf[base + i] = f2bf(o);
    if (t + 1 < T) wk_bf[base + (long long)D + i] = f2bf(o);  // wk[b,t+1]
    if (t == 0)    wk_bf[base + i] = 0;                       // wk[b,0]=0
  }
}

// ---------------------------------------------------------------------------
// Generic tiled bf16 WMMA GEMM:  C[M,N] = A(MxK) * B(KxN)
//   OPA: 0 -> A(m,k)=Asrc[m*lda+k]      1 -> A(m,k)=Asrc[k*lda+m]
//   OPB: 0 -> B(k,n)=Bsrc[k*ldb+n]      1 -> B(k,n)=Bsrc[n*ldb+k]
//   SCALEA: multiply A(.,k) by exp((Tdim-1-k)*lg) at staging (W_new path)
//   EPI: 0 store bf16
//        1 scores: c *= (t>s)? exp((t-1-s)*lg) : 0 ; store bf16
//        2 reads : c += bf(auxUs[t,n]) * exp(t*lg) ; store bf16
//        3 y     : Cf[t,n] += 0.03*c   (f32 RMW; Cf preset to x+out)
//        4 W_new : Cf = c + exp(Tdim*lg)*auxF     (f32 store)
// ---------------------------------------------------------------------------
template <int OPA, int OPB, int EPI, bool SCALEA>
__global__ __launch_bounds__(256)
void gemm_wmma(const unsigned short* __restrict__ A,    long long aBatch, int lda,
               const unsigned short* __restrict__ Bsrc, long long bBatch, int ldb,
               unsigned short* __restrict__ Cus, float* __restrict__ Cf,
               long long cBatch, int ldc,
               const unsigned short* __restrict__ auxUs, long long auxUsBatch,
               const float* __restrict__ auxF,  long long auxFBatch,
               const float* __restrict__ lgPtr, int K, int Tdim) {
  __shared__ alignas(16) unsigned short lds_a [BM][LDP];   // A: [m][k]
  __shared__ alignas(16) unsigned short lds_bt[BN][LDP];   // B transposed: [n][k]

  const int b  = blockIdx.z;
  const int m0 = blockIdx.y * BM;
  const int n0 = blockIdx.x * BN;
  const int tid  = threadIdx.x;
  const int wave = tid >> 5;
  const int lane = tid & 31;
  const int wm = (wave & 3) * 32;   // wave offset in M within macro tile
  const int wn = (wave >> 2) * 64;  // wave offset in N within macro tile
  const int lm = lane & 15;
  const int kh = lane >> 4;

  const float lg = lgPtr ? lgPtr[0] : 0.f;

  const unsigned short* Ab = A    + (long long)b * aBatch;
  const unsigned short* Bb = Bsrc + (long long)b * bBatch;

  // causal skip: scores tiles strictly above the diagonal are all zero
  if (EPI == 1 && n0 >= m0 + BM) {
    unsigned short* Cb = Cus + (long long)b * cBatch;
    for (int i = tid; i < BM * BN; i += NTHREADS) {
      int mg = m0 + i / BN, ng = n0 + i % BN;
      Cb[(long long)mg * ldc + ng] = 0;
    }
    return;
  }

  const v8f vzero = {0.f, 0.f, 0.f, 0.f, 0.f, 0.f, 0.f, 0.f};
  v8f acc[2][4];
#pragma unroll
  for (int i = 0; i < 2; ++i)
#pragma unroll
    for (int j = 0; j < 4; ++j) acc[i][j] = vzero;

  for (int k0 = 0; k0 < K; k0 += BK) {
    // ---- stage A tile [BM x BK] -> lds_a[m][k] ----
    if (OPA == 0) {
      int m  = tid >> 1;                 // 0..127
      int kk = (tid & 1) * 16;           // 0 or 16
      const unsigned short* src = Ab + (long long)(m0 + m) * lda + (k0 + kk);
      if (k0 + BK < K) __builtin_prefetch(src + BK, 0, 3);
#pragma unroll
      for (int i = 0; i < 16; ++i) lds_a[m][kk + i] = src[i];
    } else {
      int kk = tid >> 3;                 // 0..31
      int mb = (tid & 7) * 16;           // 0..112
      const unsigned short* src = Ab + (long long)(k0 + kk) * lda + (m0 + mb);
      float s = 1.f;
      if (SCALEA) s = __expf((float)(Tdim - 1 - (k0 + kk)) * lg);
#pragma unroll
      for (int i = 0; i < 16; ++i) {
        unsigned short u = src[i];
        lds_a[mb + i][kk] = SCALEA ? f2bf(bf2f(u) * s) : u;
      }
    }
    // ---- stage B tile [BK x BN] -> lds_bt[n][k] (transposed) ----
    if (OPB == 0) {
      int kk = tid >> 3;                 // 0..31
      int nb = (tid & 7) * 16;           // 0..112
      const unsigned short* src = Bb + (long long)(k0 + kk) * ldb + (n0 + nb);
#pragma unroll
      for (int i = 0; i < 16; ++i) lds_bt[nb + i][kk] = src[i];
    } else {
      int n  = tid >> 1;                 // 0..127
      int kb = (tid & 1) * 16;           // 0 or 16
      const unsigned short* src = Bb + (long long)(n0 + n) * ldb + (k0 + kb);
      if (k0 + BK < K) __builtin_prefetch(src + BK, 0, 3);
#pragma unroll
      for (int i = 0; i < 16; ++i) lds_bt[n][kb + i] = src[i];
    }
    __syncthreads();

    // ---- fragment gathers: all 16B-aligned ds_load_b128 ----
    // A (16-bit 16x32, ISA 7.12.2): lane row m=lm, K = {8kh..8kh+7, 16+8kh..+7}
    v16u afrag[2];
#pragma unroll
    for (int ti = 0; ti < 2; ++ti) {
      const int mrow = wm + ti * 16 + lm;
      v8u lo = *(const v8u*)&lds_a[mrow][8 * kh];
      v8u hi = *(const v8u*)&lds_a[mrow][16 + 8 * kh];
      afrag[ti] = __builtin_shufflevector(lo, hi, 0, 1, 2, 3, 4, 5, 6, 7,
                                          8, 9, 10, 11, 12, 13, 14, 15);
    }
    // B (16-bit 32x16): lane col n=lm, K = {8kh..8kh+7, 16+8kh..+7} (matches A)
    v16u bfrag[4];
#pragma unroll
    for (int tj = 0; tj < 4; ++tj) {
      const int ncol = wn + tj * 16 + lm;
      v8u lo = *(const v8u*)&lds_bt[ncol][8 * kh];
      v8u hi = *(const v8u*)&lds_bt[ncol][16 + 8 * kh];
      bfrag[tj] = __builtin_shufflevector(lo, hi, 0, 1, 2, 3, 4, 5, 6, 7,
                                          8, 9, 10, 11, 12, 13, 14, 15);
    }
#pragma unroll
    for (int ti = 0; ti < 2; ++ti)
#pragma unroll
      for (int tj = 0; tj < 4; ++tj)
        acc[ti][tj] = __builtin_amdgcn_wmma_f32_16x16x32_bf16(
            false, __builtin_bit_cast(v16bf, afrag[ti]),
            false, __builtin_bit_cast(v16bf, bfrag[tj]),
            (short)0, acc[ti][tj], false, false);
    __syncthreads();
  }

  // ---- epilogue: C 16x16 f32 layout: lane col n=lm; VGPR i row = i+8*kh ----
#pragma unroll
  for (int ti = 0; ti < 2; ++ti) {
#pragma unroll
    for (int tj = 0; tj < 4; ++tj) {
#pragma unroll
      for (int i = 0; i < 8; ++i) {
        int mg = m0 + wm + ti * 16 + 8 * kh + i;
        int ng = n0 + wn + tj * 16 + lm;
        float c = acc[ti][tj][i];
        if (EPI == 0) {
          Cus[(long long)b * cBatch + (long long)mg * ldc + ng] = f2bf(c);
        } else if (EPI == 1) {
          float f = (mg > ng) ? __expf((float)(mg - 1 - ng) * lg) : 0.f;
          Cus[(long long)b * cBatch + (long long)mg * ldc + ng] = f2bf(c * f);
        } else if (EPI == 2) {
          float carry = bf2f(auxUs[(long long)b * auxUsBatch +
                                   (long long)mg * ldc + ng]);
          c += carry * __expf((float)mg * lg);
          Cus[(long long)b * cBatch + (long long)mg * ldc + ng] = f2bf(c);
        } else if (EPI == 3) {
          long long idx = (long long)b * cBatch + (long long)mg * ldc + ng;
          Cf[idx] += 0.03f * c;                    // y = (x+out) + 0.03*proj
        } else {  // EPI == 4
          long long idx = (long long)b * cBatch + (long long)mg * ldc + ng;
          float wold = auxF[(long long)b * auxFBatch + (long long)mg * ldc + ng];
          Cf[idx] = c + __expf((float)Tdim * lg) * wold;
        }
      }
    }
  }
}

// ---------------------------------------------------------------------------
// Host orchestration
// ---------------------------------------------------------------------------
extern "C" void kernel_launch(void* const* d_in, const int* in_sizes, int n_in,
                              void* d_out, int out_size, void* d_ws, size_t ws_size,
                              hipStream_t stream) {
  (void)in_sizes; (void)n_in; (void)out_size; (void)ws_size;
  const int B = 4, T = 4096, D = 1024;

  const float* x     = (const float*)d_in[0];
  const float* W     = (const float*)d_in[1];
  const float* g     = (const float*)d_in[2];
  const float* Wpw   = (const float*)d_in[3];
  const float* Wpr   = (const float*)d_in[4];
  const float* decay = (const float*)d_in[5];

  float* y_out    = (float*)d_out;                        // [B,T,D]
  float* Wnew_out = y_out + (long long)B * T * D;         // [B,D,D]

  // workspace carve-out (~314 MB total)
  char* ws = (char*)d_ws;
  size_t off = 0;
  auto carve = [&](size_t bytes) -> void* {
    void* p = ws + off;
    off += (bytes + 255) & ~(size_t)255;
    return p;
  };
  float*          lg        = (float*)carve(sizeof(float));
  unsigned short* out_bf    = (unsigned short*)carve((size_t)B * T * D * 2);
  unsigned short* wk_bf     = (unsigned short*)carve((size_t)B * T * D * 2);
  unsigned short* v_bf      = (unsigned short*)carve((size_t)B * T * D * 2);
  unsigned short* carry_bf  = (unsigned short*)carve((size_t)B * T * D * 2);
  unsigned short* reads_bf  = (unsigned short*)carve((size_t)B * T * D * 2);
  unsigned short* W_bf      = (unsigned short*)carve((size_t)B * D * D * 2);
  unsigned short* Wpw_bf    = (unsigned short*)carve((size_t)D * D * 2);
  unsigned short* Wpr_bf    = (unsigned short*)carve((size_t)D * D * 2);
  unsigned short* scores_bf = (unsigned short*)carve((size_t)B * T * T * 2);

  compute_lg_kernel<<<1, 64, 0, stream>>>(decay, lg);
  {
    long long n = (long long)B * D * D;
    f32_to_bf16_kernel<<<(unsigned)((n + 255) / 256), 256, 0, stream>>>(W, W_bf, n);
    n = (long long)D * D;
    f32_to_bf16_kernel<<<(unsigned)((n + 255) / 256), 256, 0, stream>>>(Wpw, Wpw_bf, n);
    f32_to_bf16_kernel<<<(unsigned)((n + 255) / 256), 256, 0, stream>>>(Wpr, Wpr_bf, n);
  }
  rmsnorm_kernel<<<B * T, 256, 0, stream>>>(x, g, y_out, out_bf, wk_bf, T, D);

  const dim3 blk(NTHREADS);

  // 1) v = out @ Wpw^T    [T,D]x[D,D]
  gemm_wmma<0, 1, 0, false><<<dim3(D / BN, T / BM, B), blk, 0, stream>>>(
      out_bf, (long long)T * D, D, Wpw_bf, 0LL, D,
      v_bf, nullptr, (long long)T * D, D,
      nullptr, 0LL, nullptr, 0LL, lg, D, T);

  // 2) carry = out @ W[b]^T
  gemm_wmma<0, 1, 0, false><<<dim3(D / BN, T / BM, B), blk, 0, stream>>>(
      out_bf, (long long)T * D, D, W_bf, (long long)D * D, D,
      carry_bf, nullptr, (long long)T * D, D,
      nullptr, 0LL, nullptr, 0LL, lg, D, T);

  // 3) scores = (out @ wk^T) * decay-mask   [T,T], causal skip above diagonal
  gemm_wmma<0, 1, 1, false><<<dim3(T / BN, T / BM, B), blk, 0, stream>>>(
      out_bf, (long long)T * D, D, wk_bf, (long long)T * D, D,
      scores_bf, nullptr, (long long)T * T, T,
      nullptr, 0LL, nullptr, 0LL, lg, D, T);

  // 4) reads = scores @ v + carry * exp(t*lg)
  gemm_wmma<0, 0, 2, false><<<dim3(D / BN, T / BM, B), blk, 0, stream>>>(
      scores_bf, (long long)T * T, T, v_bf, (long long)T * D, D,
      reads_bf, nullptr, (long long)T * D, D,
      carry_bf, (long long)T * D, nullptr, 0LL, lg, T, T);

  // 5) y += 0.03 * (reads @ Wpr^T)   (y already holds x+out)
  gemm_wmma<0, 1, 3, false><<<dim3(D / BN, T / BM, B), blk, 0, stream>>>(
      reads_bf, (long long)T * D, D, Wpr_bf, 0LL, D,
      nullptr, y_out, (long long)T * D, D,
      nullptr, 0LL, nullptr, 0LL, lg, D, T);

  // 6) W_new = (w ⊙ v)^T @ wk + exp(T*lg)*W   [D,D]
  gemm_wmma<1, 0, 4, true><<<dim3(D / BN, D / BM, B), blk, 0, stream>>>(
      v_bf, (long long)T * D, D, wk_bf, (long long)T * D, D,
      nullptr, Wnew_out, (long long)D * D, D,
      nullptr, 0LL, W, (long long)D * D, lg, T, T);
}
